// DFTbaselined_13872744366328
// MI455X (gfx1250) — matrix operand order repaired and checked by reading, not compiled, not run on hardware
//
#include <hip/hip_runtime.h>
#include <hip/hip_bf16.h>
#include <math.h>

typedef _Float16 h16;
typedef __attribute__((ext_vector_type(16))) _Float16 v16h;
typedef __attribute__((ext_vector_type(8)))  float    v8f;

#define DI __device__ __forceinline__

constexpr int cN = 8000, cE = 128000, cG = 4;
constexpr int cC = 64, cNLM = 16, cNEL = 10, cNB = 8;
constexpr int cNK = 342;                 // 7^3 - 1 k-vectors
constexpr float cPI = 3.14159265358979f;

__constant__ int d_lmap[16] = {0,1,1,1,2,2,2,2,2,3,3,3,3,3,3,3};

// ---- workspace byte offsets (all 64B aligned) ----
constexpr size_t OFF_F    = 0;                      // feats  N*1024 f32
constexpr size_t OFF_B1   = 32768000;               // h / Ahat
constexpr size_t OFF_B2   = 65536000;               // Amsg
constexpr size_t OFF_Y    = 98304000;               // E*16 f32
constexpr size_t OFF_EF   = 106496000;              // E*8 f32
constexpr size_t OFF_ELEM = 110592000;              // N int
constexpr size_t OFF_WF   = 110624000;              // 2*220 frags * 512 h16
constexpr size_t OFF_SR   = 111074560;              // G*342 f32
constexpr size_t OFF_SI   = 111082752;

// frag-unit bases within one t (each frag = 512 halfs)
constexpr int FB_RW1 = 0, FB_RW2 = 4, FB_RW3 = 12, FB_RW4 = 20;
constexpr int FB_LU = 52, FB_LO = 84, FB_SC = 116, FB_WP = 196;
constexpr int FRAGS_PER_T = 220;

DI int laneid() { return (int)(threadIdx.x & 31u); }

DI v8f zero8() {
  v8f z;
  #pragma unroll
  for (int i = 0; i < 8; i++) z[i] = 0.f;
  return z;
}

DI v8f wmma16(v16h a, v16h b, v8f c) {
  return __builtin_amdgcn_wmma_f32_16x16x32_f16(false, a, false, b, (short)0, c, false, false);
}

// B fragment: pre-swizzled contiguous 32 lanes x 16 halfs
DI v16h bfrag(const h16* fb, int frag) {
  const h16* p = fb + (size_t)frag * 512 + laneid() * 16;
  v16h b;
  #pragma unroll
  for (int i = 0; i < 16; i++) b[i] = p[i];
  return b;
}

// A fragment from LDS row-major f16 act[16][ldk]
DI v16h afrag_lds(const h16* act, int ldk, int kt) {
  int lane = laneid(), half = lane >> 4, m = lane & 15;
  int b0 = kt * 32 + half * 8;
  v16h a;
  #pragma unroll
  for (int i = 0; i < 8; i++) {
    a[i]     = act[m * ldk + b0 + i];
    a[8 + i] = act[m * ldk + b0 + 16 + i];
  }
  return a;
}

DI void dstore_lds_silu(v8f c, h16* act, int ldn, int nt) {
  int lane = laneid(), half = lane >> 4, col = nt * 16 + (lane & 15);
  #pragma unroll
  for (int r = 0; r < 8; r++) {
    float x = c[r];
    x = x / (1.f + __expf(-x));
    act[(half * 8 + r) * ldn + col] = (h16)x;
  }
}

DI void dstore_lds_scale(v8f c, h16* act, int ldn, int nt, float s) {
  int lane = laneid(), half = lane >> 4, col = nt * 16 + (lane & 15);
  #pragma unroll
  for (int r = 0; r < 8; r++) act[(half * 8 + r) * ldn + col] = (h16)(c[r] * s);
}

// ---------------- prep kernels ----------------

__global__ void k_elem_init(const float* __restrict__ attrs, const float* __restrict__ q,
                            const float* __restrict__ pos, const int* __restrict__ batch,
                            const float* __restrict__ atomicE, int* __restrict__ elem,
                            float* __restrict__ out) {
  int n = blockIdx.x * blockDim.x + threadIdx.x;
  if (n >= cN) return;
  int best = 0; float bv = attrs[n * cNEL];
  for (int k = 1; k < cNEL; k++) { float v = attrs[n * cNEL + k]; if (v > bv) { bv = v; best = k; } }
  elem[n] = best;
  int g = batch[n];
  float qn = q[n];
  atomicAdd(&out[g], atomicE[best]);                    // e0
  atomicAdd(&out[4 + g], qn);                           // total charge
  atomicAdd(&out[8 + g * 3 + 0], qn * pos[n * 3 + 0]);  // dipole
  atomicAdd(&out[8 + g * 3 + 1], qn * pos[n * 3 + 1]);
  atomicAdd(&out[8 + g * 3 + 2], qn * pos[n * 3 + 2]);
}

__global__ void k_feats_init(const float* __restrict__ embW, const int* __restrict__ elem,
                             float* __restrict__ F) {
  int idx = blockIdx.x * blockDim.x + threadIdx.x;     // N*C threads
  if (idx >= cN * cC) return;
  int n = idx >> 6, c = idx & 63;
  float* p = F + (size_t)n * 1024 + c * 16;
  p[0] = embW[elem[n] * cC + c];
  #pragma unroll
  for (int l = 1; l < cNLM; l++) p[l] = 0.f;
}

__global__ void k_qupd(const float* __restrict__ q, const float* __restrict__ qcoef,
                       float* __restrict__ F, int t) {
  int idx = blockIdx.x * blockDim.x + threadIdx.x;
  if (idx >= cN * cC) return;
  int n = idx >> 6, c = idx & 63;
  F[(size_t)n * 1024 + c * 16] += q[n] * qcoef[t * cC + c];
}

// swizzle a KxNn row-major f32 weight into B fragments (zero-padded K)
__global__ void k_swz(const float* __restrict__ src, int K, int Nn, h16* __restrict__ dst) {
  int frag = blockIdx.x;
  int tid = threadIdx.x;                 // 512 threads
  int lane = tid >> 4, i = tid & 15;
  int ntiles = Nn >> 4;
  int kt = frag / ntiles, nt = frag % ntiles;
  int k = kt * 32 + (lane >> 4) * 8 + i + (i >= 8 ? 8 : 0);
  int n = nt * 16 + (lane & 15);
  float v = (k < K) ? src[k * Nn + n] : 0.f;
  dst[(size_t)frag * 512 + lane * 16 + i] = (h16)v;
}

// ---------------- edge geometry: Y (E x 16) and radial basis (E x 8) ----------------

__global__ void k_geom(const float* __restrict__ pos, const float* __restrict__ shifts,
                       const int* __restrict__ ei, float* __restrict__ Yb, float* __restrict__ EFb) {
  int e = blockIdx.x * blockDim.x + threadIdx.x;
  if (e >= cE) return;
  int s = ei[e], rr = ei[cE + e];
  float vx = pos[rr * 3 + 0] - pos[s * 3 + 0] + shifts[e * 3 + 0];
  float vy = pos[rr * 3 + 1] - pos[s * 3 + 1] + shifts[e * 3 + 1];
  float vz = pos[rr * 3 + 2] - pos[s * 3 + 2] + shifts[e * 3 + 2];
  float r = sqrtf(vx * vx + vy * vy + vz * vz);
  float inv = 1.f / (r + 1e-9f);
  float x = vx * inv, y = vy * inv, z = vz * inv;
  const float c3 = 1.7320508075688772f, c5 = 2.2360679774997896f, c15 = 3.872983346207417f;
  float* Yp = Yb + (size_t)e * 16;
  Yp[0] = 1.f;
  Yp[1] = c3 * x; Yp[2] = c3 * y; Yp[3] = c3 * z;
  Yp[4] = c15 * x * y; Yp[5] = c15 * y * z;
  Yp[6] = 0.5f * c5 * (3.f * z * z - 1.f);
  Yp[7] = c15 * x * z;
  Yp[8] = 0.5f * c15 * (x * x - y * y);
  Yp[9]  = 2.0916500663351889f * y * (3.f * x * x - y * y);     // sqrt(35/8)
  Yp[10] = 10.246950765959598f * x * y * z;                     // sqrt(105)
  Yp[11] = 1.6201851746019651f * y * (5.f * z * z - 1.f);       // sqrt(21/8)
  Yp[12] = 0.5f * 2.6457513110645907f * z * (5.f * z * z - 3.f);// sqrt(7)
  Yp[13] = 1.6201851746019651f * x * (5.f * z * z - 1.f);
  Yp[14] = 0.5f * 10.246950765959598f * z * (x * x - y * y);
  Yp[15] = 2.0916500663351889f * x * (x * x - 3.f * y * y);
  // radial basis
  float u = r / 5.0f;
  float fc = 0.f;
  if (u < 1.f) {
    float u5 = u * u; u5 *= u5; u5 *= u;      // u^5
    float u6 = u5 * u, u7 = u6 * u;
    fc = 1.f - 21.f * u5 + 35.f * u6 - 15.f * u7;
  }
  const float cb = 0.63245553203367587f;      // sqrt(2/5)
  float* Ep = EFb + (size_t)e * 8;
  #pragma unroll
  for (int nn = 1; nn <= cNB; nn++)
    Ep[nn - 1] = cb * __sinf((float)nn * cPI * u) * inv * fc;
}

// ---------------- per-l 64x64 GEMM tile kernel (lin_up / lin_out) ----------------

__global__ __launch_bounds__(128) void k_lin(const float* __restrict__ src, float* __restrict__ dst,
                                             const h16* __restrict__ WF, int fragbase) {
  int wave = (int)(threadIdx.x >> 5);
  int tile = blockIdx.x * 4 + wave;          // 500 node-blocks * 16 l
  int nblk = tile >> 4, l = tile & 15;
  int n0 = nblk * 16;
  const h16* fb = WF + (size_t)(fragbase + d_lmap[l] * 8) * 512;
  int lane = laneid(), half = lane >> 4, m = lane & 15;
  v8f acc[4];
  #pragma unroll
  for (int nt = 0; nt < 4; nt++) acc[nt] = zero8();
  #pragma unroll
  for (int kt = 0; kt < 2; kt++) {
    int b0 = kt * 32 + half * 8;
    v16h a;
    #pragma unroll
    for (int i = 0; i < 8; i++) {
      a[i]     = (h16)src[(size_t)(n0 + m) * 1024 + (b0 + i) * 16 + l];
      a[8 + i] = (h16)src[(size_t)(n0 + m) * 1024 + (b0 + 16 + i) * 16 + l];
    }
    #pragma unroll
    for (int nt = 0; nt < 4; nt++) acc[nt] = wmma16(a, bfrag(fb, kt * 4 + nt), acc[nt]);
  }
  #pragma unroll
  for (int nt = 0; nt < 4; nt++) {
    int col = nt * 16 + (lane & 15);
    #pragma unroll
    for (int r = 0; r < 8; r++)
      dst[(size_t)(n0 + half * 8 + r) * 1024 + col * 16 + l] = acc[nt][r];
  }
}

// ---------------- fused radial MLP (WMMA) + message scatter ----------------

__global__ __launch_bounds__(128) void k_msg(const float* __restrict__ EFb, const float* __restrict__ Yb,
                                             const float* __restrict__ h, const int* __restrict__ ei,
                                             float* __restrict__ Amsg, const h16* __restrict__ WF, int tbase) {
  __shared__ h16 sAct[4][16 * 64];
  __shared__ h16 sW[4][16 * 256];
  int wave = (int)(threadIdx.x >> 5);
  int lane = laneid(), half = lane >> 4, m = lane & 15;
  int e0 = (blockIdx.x * 4 + wave) * 16;
  h16* act = sAct[wave];
  h16* wbuf = sW[wave];
  const h16* fb1 = WF + (size_t)(tbase + FB_RW1) * 512;
  const h16* fb2 = WF + (size_t)(tbase + FB_RW2) * 512;
  const h16* fb3 = WF + (size_t)(tbase + FB_RW3) * 512;
  const h16* fb4 = WF + (size_t)(tbase + FB_RW4) * 512;

  // layer 1: (16x8) @ (8x64), K zero-padded to 32
  {
    v16h a;
    #pragma unroll
    for (int i = 0; i < 16; i++) a[i] = (h16)0.f;
    if (half == 0) {
      #pragma unroll
      for (int i = 0; i < 8; i++) a[i] = (h16)EFb[(size_t)(e0 + m) * 8 + i];
    }
    #pragma unroll
    for (int nt = 0; nt < 4; nt++) {
      v8f c = wmma16(a, bfrag(fb1, nt), zero8());
      dstore_lds_silu(c, act, 64, nt);
    }
  }
  __syncthreads();
  // layers 2,3: (16x64)@(64x64)
  for (int layer = 0; layer < 2; layer++) {
    const h16* fb = layer ? fb3 : fb2;
    v8f acc[4];
    #pragma unroll
    for (int nt = 0; nt < 4; nt++) acc[nt] = zero8();
    #pragma unroll
    for (int kt = 0; kt < 2; kt++) {
      v16h a = afrag_lds(act, 64, kt);
      #pragma unroll
      for (int nt = 0; nt < 4; nt++) acc[nt] = wmma16(a, bfrag(fb, kt * 4 + nt), acc[nt]);
    }
    __syncthreads();
    #pragma unroll
    for (int nt = 0; nt < 4; nt++) dstore_lds_silu(acc[nt], act, 64, nt);
    __syncthreads();
  }
  // layer 4: (16x64)@(64x256), scale by 1/AVG_NEIGH, no activation
  {
    v8f acc[16];
    #pragma unroll
    for (int nt = 0; nt < 16; nt++) acc[nt] = zero8();
    #pragma unroll
    for (int kt = 0; kt < 2; kt++) {
      v16h a = afrag_lds(act, 64, kt);
      #pragma unroll
      for (int nt = 0; nt < 16; nt++) acc[nt] = wmma16(a, bfrag(fb4, kt * 16 + nt), acc[nt]);
    }
    #pragma unroll
    for (int nt = 0; nt < 16; nt++) dstore_lds_scale(acc[nt], wbuf, 256, nt, 1.f / 16.f);
  }
  __syncthreads();

  // message phase: m = w * Y * (hj . Y), scatter-add into Amsg[recv]
  for (int el = 0; el < 16; el++) {
    int e = e0 + el;
    int snd = ei[e], rcv = ei[cE + e];
    float Yl[16];
    #pragma unroll
    for (int l = 0; l < 16; l++) Yl[l] = Yb[(size_t)e * 16 + l];
    const float* hj = h + (size_t)snd * 1024;
    float* Ar = Amsg + (size_t)rcv * 1024;
    #pragma unroll
    for (int cc = 0; cc < 2; cc++) {
      int c = lane + cc * 32;
      float se = 0.f;
      #pragma unroll
      for (int l = 0; l < 16; l++) se += hj[c * 16 + l] * Yl[l];
      const h16* wrow = wbuf + el * 256 + c * 4;
      float w4[4];
      #pragma unroll
      for (int j = 0; j < 4; j++) w4[j] = (float)wrow[j];
      #pragma unroll
      for (int l = 0; l < 16; l++)
        atomicAdd(&Ar[c * 16 + l], w4[d_lmap[l]] * Yl[l] * se);
    }
  }
}

// ---------------- per-node: Wp polynomial + W_sc skip, all WMMA ----------------

__global__ __launch_bounds__(128) void k_node(const float* __restrict__ Ahat, float* __restrict__ F,
                                              const int* __restrict__ elem, const h16* __restrict__ WF,
                                              int tbase) {
  __shared__ float sS1[4][64];
  int wave = (int)(threadIdx.x >> 5);
  int lane = laneid(), half = lane >> 4, m = lane & 15;
  int n = blockIdx.x * 4 + wave;
  const float* An = Ahat + (size_t)n * 1024;
  float* Fn = F + (size_t)n * 1024;
  #pragma unroll
  for (int cc = 0; cc < 2; cc++) {
    int c = lane + cc * 32;
    float s = 0.f;
    #pragma unroll
    for (int l = 0; l < 16; l++) { float v = An[c * 16 + l]; s += v * v; }
    sS1[wave][c] = s;
  }
  __syncthreads();
  int e = elem[n];
  const h16* fwp0 = WF + (size_t)(tbase + FB_WP + 0) * 512;
  const h16* fwp1 = WF + (size_t)(tbase + FB_WP + 8) * 512;
  const h16* fwp2 = WF + (size_t)(tbase + FB_WP + 16) * 512;
  const h16* fsc  = WF + (size_t)(tbase + FB_SC + e * 8) * 512;
  v8f acc[4];
  #pragma unroll
  for (int nt = 0; nt < 4; nt++) acc[nt] = zero8();
  #pragma unroll
  for (int kt = 0; kt < 2; kt++) {
    int b0 = kt * 32 + half * 8;
    v16h a0, a1, a2, af;
    #pragma unroll
    for (int i = 0; i < 8; i++) {
      int k1 = b0 + i, k2 = b0 + 16 + i;
      float v1 = An[k1 * 16 + m], v2 = An[k2 * 16 + m];
      float s1 = sS1[wave][k1], s2 = sS1[wave][k2];
      a0[i] = (h16)v1;               a0[8 + i] = (h16)v2;
      a1[i] = (h16)(v1 * s1);        a1[8 + i] = (h16)(v2 * s2);
      a2[i] = (h16)(v1 * s1 * s1);   a2[8 + i] = (h16)(v2 * s2 * s2);
      af[i] = (h16)Fn[k1 * 16 + m];  af[8 + i] = (h16)Fn[k2 * 16 + m];
    }
    #pragma unroll
    for (int nt = 0; nt < 4; nt++) {
      acc[nt] = wmma16(a0, bfrag(fwp0, kt * 4 + nt), acc[nt]);
      acc[nt] = wmma16(a1, bfrag(fwp1, kt * 4 + nt), acc[nt]);
      acc[nt] = wmma16(a2, bfrag(fwp2, kt * 4 + nt), acc[nt]);
      acc[nt] = wmma16(af, bfrag(fsc,  kt * 4 + nt), acc[nt]);
    }
  }
  // D rows = l, cols = d  ->  feats[n, d, l]
  #pragma unroll
  for (int nt = 0; nt < 4; nt++) {
    int col = nt * 16 + (lane & 15);
    #pragma unroll
    for (int r = 0; r < 8; r++)
      Fn[col * 16 + (half * 8 + r)] = acc[nt][r];
  }
}

// ---------------- readout ----------------

__global__ void k_read(const float* __restrict__ F, const float* __restrict__ q,
                       const int* __restrict__ batch, const float* __restrict__ Wr0,
                       const float* __restrict__ Wa, const float* __restrict__ Wb,
                       const float* __restrict__ Wq, float* __restrict__ out, int t) {
  int n = blockIdx.x * blockDim.x + threadIdx.x;
  if (n >= cN) return;
  float f0[64];
  #pragma unroll
  for (int c = 0; c < 64; c++) f0[c] = F[(size_t)n * 1024 + c * 16];
  float ne = 0.f;
  if (t == 1) {
    for (int j = 0; j < 16; j++) {
      float hsum = 0.f;
      #pragma unroll
      for (int c = 0; c < 64; c++) hsum += f0[c] * Wa[c * 16 + j];
      hsum = hsum / (1.f + __expf(-hsum));
      ne += hsum * Wb[j];
    }
  } else {
    #pragma unroll
    for (int c = 0; c < 64; c++) ne += f0[c] * Wr0[c];
  }
  float qq = 0.f;
  #pragma unroll
  for (int c = 0; c < 64; c++) qq += f0[c] * Wq[t * 64 + c];
  ne += q[n] * qq;
  atomicAdd(&out[batch[n]], ne);
}

// ---------------- Ewald reciprocal sum ----------------

DI void kvec_decode(int k, const float* rc, float& kx, float& ky, float& kz) {
  int idx = (k < 171) ? k : k + 1;
  float na = (float)(idx / 49 - 3);
  float nb = (float)((idx / 7) % 7 - 3);
  float nc = (float)(idx % 7 - 3);
  kx = 2.f * cPI * (na * rc[0] + nb * rc[3] + nc * rc[6]);
  ky = 2.f * cPI * (na * rc[1] + nb * rc[4] + nc * rc[7]);
  kz = 2.f * cPI * (na * rc[2] + nb * rc[5] + nc * rc[8]);
}

__global__ void k_sf(const float* __restrict__ pos, const float* __restrict__ q,
                     const int* __restrict__ batch, const float* __restrict__ rcell,
                     float* __restrict__ Sr, float* __restrict__ Si) {
  __shared__ float rs[256], ri[256];
  int gk = blockIdx.x;
  int g = gk / cNK, k = gk % cNK;
  float kx, ky, kz;
  kvec_decode(k, rcell + g * 9, kx, ky, kz);
  float sr = 0.f, si = 0.f;
  for (int n = threadIdx.x; n < cN; n += blockDim.x) {
    if (batch[n] != g) continue;
    float ph = pos[n * 3] * kx + pos[n * 3 + 1] * ky + pos[n * 3 + 2] * kz;
    float qn = q[n];
    sr += qn * __cosf(ph);
    si += qn * __sinf(ph);
  }
  rs[threadIdx.x] = sr; ri[threadIdx.x] = si;
  __syncthreads();
  for (int st = 128; st > 0; st >>= 1) {
    if ((int)threadIdx.x < st) { rs[threadIdx.x] += rs[threadIdx.x + st]; ri[threadIdx.x] += ri[threadIdx.x + st]; }
    __syncthreads();
  }
  if (threadIdx.x == 0) { Sr[gk] = rs[0]; Si[gk] = ri[0]; }
}

__global__ void k_ewald(const float* __restrict__ Sr, const float* __restrict__ Si,
                        const float* __restrict__ rcell, const float* __restrict__ vol,
                        float* __restrict__ out) {
  __shared__ float red[256];
  int g = blockIdx.x;
  float part = 0.f;
  for (int k = threadIdx.x; k < cNK; k += blockDim.x) {
    float kx, ky, kz;
    kvec_decode(k, rcell + g * 9, kx, ky, kz);
    float k2 = kx * kx + ky * ky + kz * kz;
    float filt = __expf(-0.5f * k2) / k2;
    float sr = Sr[g * cNK + k], si = Si[g * cNK + k];
    part += filt * (sr * sr + si * si);
  }
  red[threadIdx.x] = part;
  __syncthreads();
  for (int st = 128; st > 0; st >>= 1) {
    if ((int)threadIdx.x < st) red[threadIdx.x] += red[threadIdx.x + st];
    __syncthreads();
  }
  if (threadIdx.x == 0) atomicAdd(&out[g], 2.f * cPI / vol[g] * red[0]);
}

// ---------------- host launch ----------------

extern "C" void kernel_launch(void* const* d_in, const int* in_sizes, int n_in,
                              void* d_out, int out_size, void* d_ws, size_t ws_size,
                              hipStream_t stream) {
  (void)in_sizes; (void)n_in; (void)out_size; (void)ws_size;
  const float* node_attrs = (const float*)d_in[0];
  const float* positions  = (const float*)d_in[1];
  const float* shifts     = (const float*)d_in[2];
  const float* q          = (const float*)d_in[3];
  const float* rcell      = (const float*)d_in[5];
  const float* volume     = (const float*)d_in[6];
  const int*   edge_index = (const int*)d_in[7];
  const int*   batch      = (const int*)d_in[8];
  const float* emb_W      = (const float*)d_in[9];
  const float* q_coef     = (const float*)d_in[10];
  const float* rad_W1     = (const float*)d_in[11];
  const float* rad_W2     = (const float*)d_in[12];
  const float* rad_W3     = (const float*)d_in[13];
  const float* rad_W4     = (const float*)d_in[14];
  const float* lin_up     = (const float*)d_in[15];
  const float* lin_out    = (const float*)d_in[16];
  const float* W_sc       = (const float*)d_in[17];
  const float* Wp         = (const float*)d_in[18];
  const float* W_read0    = (const float*)d_in[19];
  const float* Wread1_a   = (const float*)d_in[20];
  const float* Wread1_b   = (const float*)d_in[21];
  const float* Wq_read    = (const float*)d_in[22];
  const float* atomic_E   = (const float*)d_in[23];

  char* ws = (char*)d_ws;
  float* F    = (float*)(ws + OFF_F);
  float* B1   = (float*)(ws + OFF_B1);
  float* B2   = (float*)(ws + OFF_B2);
  float* Yb   = (float*)(ws + OFF_Y);
  float* EFb  = (float*)(ws + OFF_EF);
  int*   elem = (int*)(ws + OFF_ELEM);
  h16*   WF   = (h16*)(ws + OFF_WF);
  float* Sr   = (float*)(ws + OFF_SR);
  float* Si   = (float*)(ws + OFF_SI);
  float* out  = (float*)d_out;

  (void)hipMemsetAsync(d_out, 0, 20 * sizeof(float), stream);

  k_elem_init<<<(cN + 255) / 256, 256, 0, stream>>>(node_attrs, q, positions, batch, atomic_E, elem, out);
  k_feats_init<<<(cN * cC) / 256, 256, 0, stream>>>(emb_W, elem, F);

  // swizzle all weights into B fragments (f16), both layers
  for (int t = 0; t < 2; t++) {
    int tb = t * FRAGS_PER_T;
    k_swz<<<4, 512, 0, stream>>>(rad_W1 + t * 8 * 64,   8,  64,  WF + (size_t)(tb + FB_RW1) * 512);
    k_swz<<<8, 512, 0, stream>>>(rad_W2 + t * 4096,     64, 64,  WF + (size_t)(tb + FB_RW2) * 512);
    k_swz<<<8, 512, 0, stream>>>(rad_W3 + t * 4096,     64, 64,  WF + (size_t)(tb + FB_RW3) * 512);
    k_swz<<<32, 512, 0, stream>>>(rad_W4 + t * 64 * 256, 64, 256, WF + (size_t)(tb + FB_RW4) * 512);
    for (int l = 0; l < 4; l++) {
      k_swz<<<8, 512, 0, stream>>>(lin_up  + (size_t)(t * 4 + l) * 4096, 64, 64, WF + (size_t)(tb + FB_LU + l * 8) * 512);
      k_swz<<<8, 512, 0, stream>>>(lin_out + (size_t)(t * 4 + l) * 4096, 64, 64, WF + (size_t)(tb + FB_LO + l * 8) * 512);
    }
    for (int k = 0; k < cNEL; k++)
      k_swz<<<8, 512, 0, stream>>>(W_sc + (size_t)(t * cNEL + k) * 4096, 64, 64, WF + (size_t)(tb + FB_SC + k * 8) * 512);
    for (int j = 0; j < 3; j++)
      k_swz<<<8, 512, 0, stream>>>(Wp + (size_t)(t * 3 + j) * 4096, 64, 64, WF + (size_t)(tb + FB_WP + j * 8) * 512);
  }

  k_geom<<<(cE + 255) / 256, 256, 0, stream>>>(positions, shifts, edge_index, Yb, EFb);

  for (int t = 0; t < 2; t++) {
    int tb = t * FRAGS_PER_T;
    k_qupd<<<(cN * cC) / 256, 256, 0, stream>>>(q, q_coef, F, t);
    k_lin<<<2000, 128, 0, stream>>>(F, B1, WF, tb + FB_LU);          // h = lin_up per-l
    (void)hipMemsetAsync(B2, 0, (size_t)cN * 1024 * sizeof(float), stream);
    k_msg<<<2000, 128, 0, stream>>>(EFb, Yb, B1, edge_index, B2, WF, tb);
    k_lin<<<2000, 128, 0, stream>>>(B2, B1, WF, tb + FB_LO);         // Ahat = lin_out per-l
    k_node<<<2000, 128, 0, stream>>>(B1, F, elem, WF, tb);           // Wp poly + W_sc skip
    k_read<<<(cN + 255) / 256, 256, 0, stream>>>(F, q, batch, W_read0, Wread1_a, Wread1_b, Wq_read, out, t);
  }

  k_sf<<<cG * cNK, 256, 0, stream>>>(positions, q, batch, rcell, Sr, Si);
  k_ewald<<<cG, 256, 0, stream>>>(Sr, Si, rcell, volume, out);
}
